// AdaptiveGraphSpatialAttention_71270687309843
// MI455X (gfx1250) — matrix operand order repaired and checked by reading, not compiled, not run on hardware
//
#include <hip/hip_runtime.h>

#define DD 17
#define MM 512
#define HH 8
#define DK 64
#define NTOK 8192
#define SCS 20   // scores row stride (floats), >= DD, keeps LDS under 160KB

typedef __attribute__((ext_vector_type(16))) __bf16        v16bf;
typedef __attribute__((ext_vector_type(8)))  float         v8f;
typedef __attribute__((ext_vector_type(4)))  unsigned int  u32x4;

union FragB { v16bf bf; u32x4 q[2]; };

__device__ inline unsigned short f2bf(float f) {
  union { float f; unsigned u; } v; v.f = f;
  unsigned r = v.u + 0x7FFFu + ((v.u >> 16) & 1u);  // round-to-nearest-even
  return (unsigned short)(r >> 16);
}

// ---------------------------------------------------------------------------
// Pre-pack a 512x512 fp32 weight (nn.Linear layout: out col n uses W row n)
// into bf16 WMMA B-fragments. Layout per (ntile, ktile):
//   lane L: N = ntile*16 + (L&15); K base = ktile*32 + (L>=16 ? 16 : 0);
//   dword j packs K = base+2j, base+2j+1.
// ---------------------------------------------------------------------------
__global__ void pack_w_kernel(const float* __restrict__ W,
                              unsigned int* __restrict__ out) {
  int t = blockIdx.x * blockDim.x + threadIdx.x;   // 0 .. 32*16*32-1
  if (t >= 32 * 16 * 32) return;
  int lane  = t & 31;
  int kt    = (t >> 5) & 15;
  int ntile = t >> 9;
  int n     = ntile * 16 + (lane & 15);
  int mbase = kt * 32 + ((lane & 16) ? 16 : 0);
  unsigned int* dst = out + (size_t)t * 8;
  const float* wr = W + (size_t)n * MM;
#pragma unroll
  for (int j = 0; j < 8; ++j) {
    unsigned lo = f2bf(wr[mbase + 2 * j]);
    unsigned hi = f2bf(wr[mbase + 2 * j + 1]);
    dst[j] = lo | (hi << 16);
  }
}

// ---------------------------------------------------------------------------
// Main fused kernel: one token (17x512) per 256-thread block (8 wave32s).
// LDS plan (158,336 B total -> 2 workgroups per 320KB WGP):
//   [     0, 32768)  XC : 32x512 bf16, X then Ctx
//   [ 32768, 65536)  Qs : 32x512 bf16   --+  OF (17x512 f32, 34816 B) aliases
//   [ 65536, 98304)  Ks : 32x512 bf16   --+  [32768, 67584) after attention
//   [ 98304,131072)  Vt : [512][32] bf16 (transposed V)
//   [131072,141952)  SC : [8][17][SCS] f32 scores
//   [141952,158336)  AT : [8][32][32] bf16 attn
// ---------------------------------------------------------------------------
__global__ __launch_bounds__(256, 2)
void agsa_kernel(const float* __restrict__ x, const float* __restrict__ P,
                 const float* __restrict__ bo, const float* __restrict__ gamma,
                 const float* __restrict__ beta,
                 const unsigned int* __restrict__ wq,
                 const unsigned int* __restrict__ wk,
                 const unsigned int* __restrict__ wv,
                 const unsigned int* __restrict__ wo,
                 float* __restrict__ out) {
  extern __shared__ char smem[];
  unsigned short* XC = (unsigned short*)(smem);
  unsigned short* Qs = (unsigned short*)(smem + 32768);
  unsigned short* Ks = (unsigned short*)(smem + 65536);
  unsigned short* Vt = (unsigned short*)(smem + 98304);
  float*          SC = (float*)(smem + 131072);
  unsigned short* AT = (unsigned short*)(smem + 141952);
  float*          OF = (float*)(smem + 32768);   // aliases Qs (+2KB of Ks)

  const int tid  = threadIdx.x;
  const int w    = tid >> 5;       // wave id 0..7
  const int lane = tid & 31;
  const int hi   = (lane >> 4) & 1;
  const int ln   = lane & 15;
  const float* xg = x + (size_t)blockIdx.x * (DD * MM);

  auto WMMA = [](v16bf a, v16bf b, v8f c) -> v8f {
    return __builtin_amdgcn_wmma_f32_16x16x32_bf16(false, a, false, b,
                                                   (short)0, c, false, false);
  };
  // A fragment (16x32 bf16) from a row-major bf16 LDS buffer, row stride 512.
  // lanes 0-15: K {0-15}; lanes 16-31: K {16-31}.
  auto loadA = [&](const unsigned short* base, int row, int kbase) -> v16bf {
    FragB f;
    const unsigned short* p = base + row * MM + kbase + (hi ? 16 : 0);
    f.q[0] = *(const u32x4*)(p);
    f.q[1] = *(const u32x4*)(p + 8);
    return f.bf;
  };
  // B fragment from prepacked global weights: 32 contiguous bytes per lane.
  auto loadB = [&](const unsigned int* wpk, int ntile, int kt) -> v16bf {
    FragB f;
    const unsigned int* p = wpk + ((((ntile << 4) + kt) * 32) + lane) * 8;
    f.q[0] = *(const u32x4*)(p);
    f.q[1] = *(const u32x4*)(p + 4);
    return f.bf;
  };
  // B fragment (32x16) from row-major LDS: lane column N = nbase+ln,
  // 16 contiguous K values starting at kbase (+16 for hi half-wave).
  auto loadB_rows = [&](const unsigned short* base, int nbase, int kbase) -> v16bf {
    FragB f;
    const unsigned short* p = base + (nbase + ln) * MM + kbase + (hi ? 16 : 0);
    f.q[0] = *(const u32x4*)(p);
    f.q[1] = *(const u32x4*)(p + 8);
    return f.bf;
  };

  // ---- Phase 1: x -> bf16 LDS, zero-pad rows 17..31 ----
  for (int i = tid; i < 32 * MM; i += 256) {
    int r = i >> 9;
    XC[i] = (r < DD) ? f2bf(xg[i]) : (unsigned short)0;
  }
  __syncthreads();

  // ---- Phase 2: Q/K/V projections (y = x @ W^T), WMMA bf16->f32 ----
  for (int nt = w; nt < 32; nt += 8) {
    for (int p = 0; p < 3; ++p) {
      const unsigned int* wpk = (p == 0) ? wq : (p == 1) ? wk : wv;
      v8f a0 = {}, a1 = {};
      for (int kt = 0; kt < 16; ++kt) {
        if (kt + 1 < 16)  // WGP-scope prefetch of next B fragment
          __builtin_prefetch(wpk + ((((nt << 4) + kt + 1) * 32) + lane) * 8, 0, 3);
        v16bf bfrag = loadB(wpk, nt, kt);
        v16bf x0 = loadA(XC, ln, kt * 32);
        v16bf x1 = loadA(XC, 16 + ln, kt * 32);
        a0 = WMMA(x0, bfrag, a0);
        a1 = WMMA(x1, bfrag, a1);
      }
      int col = nt * 16 + ln;
      if (p < 2) {
        unsigned short* dst = (p == 0) ? Qs : Ks;
#pragma unroll
        for (int r = 0; r < 8; ++r) {
          int row0 = r + (hi ? 8 : 0);
          dst[row0 * MM + col]        = f2bf(a0[r]);
          dst[(16 + row0) * MM + col] = f2bf(a1[r]);
        }
      } else {  // V stored transposed: Vt[feature][e]
#pragma unroll
        for (int r = 0; r < 8; ++r) {
          int row0 = r + (hi ? 8 : 0);
          Vt[col * 32 + row0]      = f2bf(a0[r]);
          Vt[col * 32 + 16 + row0] = f2bf(a1[r]);
        }
      }
    }
  }
  __syncthreads();

  // ---- Phase 3a: scores = Q K^T / sqrt(dk) + P  (one head per wave) ----
  {
    const int head = w;
    const float scl = 0.125f;  // 1/sqrt(64)
    for (int mt = 0; mt < 2; ++mt) {
      for (int nt = 0; nt < 2; ++nt) {
        v8f acc = {};
        for (int ks = 0; ks < 2; ++ks) {
          int kbase = head * DK + ks * 32;
          v16bf aq = loadA(Qs, mt * 16 + ln, kbase);
          v16bf bk = loadB_rows(Ks, nt * 16, kbase);
          acc = WMMA(aq, bk, acc);
        }
        int e = nt * 16 + ln;
#pragma unroll
        for (int r = 0; r < 8; ++r) {
          int row = mt * 16 + r + (hi ? 8 : 0);
          if (row < DD && e < DD)
            SC[(head * DD + row) * SCS + e] = acc[r] * scl + P[row * DD + e];
        }
      }
    }
  }
  __syncthreads();

  // ---- Phase 3b: softmax over e (17 valid), write bf16 attn, zero pad ----
  if (tid < HH * DD) {                    // 136 threads: one row each
    int h = tid / DD, r = tid % DD;
    const float* srow = SC + (h * DD + r) * SCS;
    float m = -1e30f;
    for (int e = 0; e < DD; ++e) m = fmaxf(m, srow[e]);
    float s = 0.f, ev[DD];
    for (int e = 0; e < DD; ++e) { ev[e] = __expf(srow[e] - m); s += ev[e]; }
    float inv = 1.f / s;
    unsigned short* arow = AT + (h * 32 + r) * 32;
    for (int e = 0; e < DD; ++e) arow[e] = f2bf(ev[e] * inv);
    for (int e = DD; e < 32; ++e) arow[e] = 0;
  } else {                                // 120 threads: zero pad rows 17..31
    int idx = tid - HH * DD;
    int h = idx / (32 - DD);
    int r = DD + idx % (32 - DD);
    unsigned short* arow = AT + (h * 32 + r) * 32;
    for (int e = 0; e < 32; ++e) arow[e] = 0;
  }
  __syncthreads();

  // ---- Phase 3c: ctx = attn @ V (K dim = e padded to 32; 1 WMMA per tile) ----
  {
    const int head = w;
    for (int mt = 0; mt < 2; ++mt) {
      for (int nt = 0; nt < 4; ++nt) {
        FragB fa;
        const unsigned short* pa =
            AT + ((head << 5) + mt * 16 + ln) * 32 + (hi ? 16 : 0);
        fa.q[0] = *(const u32x4*)(pa);
        fa.q[1] = *(const u32x4*)(pa + 8);
        int col = head * DK + nt * 16 + ln;
        FragB fb;
        const unsigned short* pb = Vt + col * 32 + (hi ? 16 : 0);
        fb.q[0] = *(const u32x4*)(pb);
        fb.q[1] = *(const u32x4*)(pb + 8);
        v8f acc = {};
        acc = WMMA(fa.bf, fb.bf, acc);
#pragma unroll
        for (int r = 0; r < 8; ++r) {
          int row = mt * 16 + r + (hi ? 8 : 0);
          XC[row * MM + col] = f2bf(acc[r]);   // Ctx reuses X buffer
        }
      }
    }
  }
  __syncthreads();

  // ---- Phase 4: out = ctx @ Wo^T + bo + residual (OF aliases dead Q/K) ----
  for (int nt = w; nt < 32; nt += 8) {
    v8f a0 = {}, a1 = {};
    for (int kt = 0; kt < 16; ++kt) {
      if (kt + 1 < 16)
        __builtin_prefetch(wo + ((((nt << 4) + kt + 1) * 32) + lane) * 8, 0, 3);
      v16bf bfrag = loadB(wo, nt, kt);
      v16bf c0 = loadA(XC, ln, kt * 32);
      v16bf c1 = loadA(XC, 16 + ln, kt * 32);
      a0 = WMMA(c0, bfrag, a0);
      a1 = WMMA(c1, bfrag, a1);
    }
    int col = nt * 16 + ln;
    float bias = bo[col];
#pragma unroll
    for (int r = 0; r < 8; ++r) {
      int row0 = r + (hi ? 8 : 0);
      OF[row0 * MM + col] = a0[r] + bias + xg[row0 * MM + col];
      int row1 = 16 + row0;
      if (row1 < DD) OF[row1 * MM + col] = a1[r] + bias + xg[row1 * MM + col];
    }
  }
  __syncthreads();

  // ---- Phase 5: LayerNorm over M=512 per row, wave32 reductions ----
  float* og = out + (size_t)blockIdx.x * (DD * MM);
  for (int row = w; row < DD; row += 8) {
    const float* orow = OF + row * MM;
    float s = 0.f, ss = 0.f;
    for (int c = lane; c < MM; c += 32) {
      float v = orow[c];
      s += v; ss += v * v;
    }
#pragma unroll
    for (int off = 16; off > 0; off >>= 1) {
      s  += __shfl_xor(s, off, 32);
      ss += __shfl_xor(ss, off, 32);
    }
    float mean = s * (1.f / MM);
    float var  = ss * (1.f / MM) - mean * mean;
    float rstd = rsqrtf(var + 1e-5f);
    for (int c = lane; c < MM; c += 32)
      og[row * MM + c] = (orow[c] - mean) * rstd * gamma[c] + beta[c];
  }
}

// ---------------------------------------------------------------------------
extern "C" void kernel_launch(void* const* d_in, const int* in_sizes, int n_in,
                              void* d_out, int out_size, void* d_ws, size_t ws_size,
                              hipStream_t stream) {
  const float* x     = (const float*)d_in[0];
  const float* P     = (const float*)d_in[1];
  const float* Wq    = (const float*)d_in[2];
  const float* Wk    = (const float*)d_in[3];
  const float* Wv    = (const float*)d_in[4];
  const float* Wo    = (const float*)d_in[5];
  const float* bo    = (const float*)d_in[6];
  const float* gamma = (const float*)d_in[7];
  const float* beta  = (const float*)d_in[8];

  unsigned int* wq_pk = (unsigned int*)d_ws;     // each packed weight: 512 KB
  unsigned int* wk_pk = wq_pk + 131072;
  unsigned int* wv_pk = wk_pk + 131072;
  unsigned int* wo_pk = wv_pk + 131072;

  pack_w_kernel<<<dim3(64), dim3(256), 0, stream>>>(Wq, wq_pk);
  pack_w_kernel<<<dim3(64), dim3(256), 0, stream>>>(Wk, wk_pk);
  pack_w_kernel<<<dim3(64), dim3(256), 0, stream>>>(Wv, wv_pk);
  pack_w_kernel<<<dim3(64), dim3(256), 0, stream>>>(Wo, wo_pk);

  size_t smem = 158336;  // fits 2 workgroups per 320KB WGP
  agsa_kernel<<<dim3(NTOK), dim3(256), smem, stream>>>(
      x, P, bo, gamma, beta, wq_pk, wk_pk, wv_pk, wo_pk, (float*)d_out);
}
